// StereoCrossAttention_30099130810945
// MI455X (gfx1250) — matrix operand order repaired and checked by reading, not compiled, not run on hardware
//
#include <hip/hip_runtime.h>

// ---------------------------------------------------------------------------
// StereoCrossAttention for MI455X (gfx1250).
// All contractions via v_wmma_f32_16x16x32_bf16 (fp32 accumulate).
// All WMMA operands staged in *fragment-major* order so every fragment load
// is two ds_load_b128 (lane's 16 bf16 = 32 contiguous bytes).
// ---------------------------------------------------------------------------

#define B_   8
#define C_   384
#define NH_  4
#define HD_  96
#define H_   64
#define W_   128
#define HW_  (H_ * W_)            // 8192
#define NPOS (B_ * HW_)           // 65536
#define NQ   ((size_t)B_ * NH_ * H_ * W_ * HD_)   // 25165824 elements
#define NX   ((size_t)B_ * C_ * HW_)              // 25165824 elements

typedef __attribute__((ext_vector_type(16))) __bf16 v16bf;
typedef __attribute__((ext_vector_type(8)))  float  v8f;

__device__ __forceinline__ unsigned short f2bf(float f) {
  unsigned u = __float_as_uint(f);
  u += 0x7fffu + ((u >> 16) & 1u);          // round-to-nearest-even
  return (unsigned short)(u >> 16);
}
__device__ __forceinline__ float bf2f(unsigned short h) {
  return __uint_as_float(((unsigned)h) << 16);
}
__device__ __forceinline__ unsigned pack2(float a, float b) {
  return (unsigned)f2bf(a) | ((unsigned)f2bf(b) << 16);
}

// Fragment-major element offsets within a 512-ushort fragment block.
// A fragment (16 M x 32 K): lane = (m) + 16*(k>>4), elem = k&15
__device__ __forceinline__ int a_off(int m, int k) {
  return ((m & 15) + 16 * ((k >> 4) & 1)) * 16 + (k & 15);
}
// B fragment (32 K x 16 N): lane = (n) + 16*(k>>4), elem = k&15
__device__ __forceinline__ int b_off(int k, int n) {
  return ((n & 15) + 16 * ((k >> 4) & 1)) * 16 + (k & 15);
}

// Load one fragment: lane's 32 contiguous bytes -> 2x ds_load_b128.
__device__ __forceinline__ v16bf frag_ld(const unsigned short* s, int lane) {
  union { v16bf v; uint4 q[2]; } f;
  const uint4* p = (const uint4*)(s + lane * 16);
  f.q[0] = p[0];
  f.q[1] = p[1];
  return f.v;
}

__device__ __forceinline__ v8f wmma_bf16(v16bf a, v16bf b, v8f c) {
  return __builtin_amdgcn_wmma_f32_16x16x32_bf16(false, a, false, b, (short)0, c,
                                                 false, false);
}

// Global fragment-major indices for q/k/v (per (b,nh,h) row of 12288 elements).
// q: A-frags   [strip=w>>4 (8)][kc=hd>>5 (3)][512]
__device__ __forceinline__ size_t q_idx(size_t base, int w, int hd) {
  return base + (size_t)((w >> 4) * 1536 + (hd >> 5) * 512 + a_off(w & 15, hd & 31));
}
// k: B-frags for S (n = w', K = hd)  [t=w'>>4 (8)][kc=hd>>5 (3)][512]
__device__ __forceinline__ size_t k_idx(size_t base, int w, int hd) {
  return base + (size_t)(((w >> 4) * 3 + (hd >> 5)) * 512 + b_off(hd & 31, w & 15));
}
// v: B-frags for P*V (n = hd, K = w') [t=hd>>4 (6)][kc=w'>>5 (4)][512]
__device__ __forceinline__ size_t v_idx(size_t base, int w, int hd) {
  return base + (size_t)(((hd >> 4) * 4 + (w >> 5)) * 512 + b_off(w & 31, hd & 15));
}

// ---------------------------------------------------------------------------
// Kernel 1: QKV projection, tile 32 co x 64 pos, K=384.
// MODE: 0 -> q layout, 1 -> k layout, 2 -> v layout.
// ---------------------------------------------------------------------------
template <int MODE>
__global__ __launch_bounds__(256) void k_proj(const float* __restrict__ x,
                                              const float* __restrict__ wmat,
                                              const float* __restrict__ bias,
                                              unsigned short* __restrict__ y) {
  __shared__ __align__(16) unsigned short As[2 * 512];
  __shared__ __align__(16) unsigned short Bs[4 * 512];
  const int tid = threadIdx.x, lane = tid & 31, wid = tid >> 5;
  const int msub = wid >> 2, nsub = wid & 3;
  const int pos0 = blockIdx.x * 64;
  const int co0  = blockIdx.y * 32;
  const int b = pos0 >> 13, rem = pos0 & 8191, h = rem >> 7, w0 = rem & 127;
  const float* xbase = x + (size_t)b * C_ * HW_ + h * W_ + w0;

  const int ar = tid >> 3, ac = (tid & 7) * 4;     // A: row, 4 consecutive k
  const int kq = tid >> 6, ng = tid & 63;          // B: 8 k-rows, position

  v8f acc = {};
  for (int k0 = 0; k0 < C_; k0 += 32) {
    {   // A tile (weights): uint2 store into fragment-major LDS
      const float* src = wmat + (size_t)(co0 + ar) * C_ + k0 + ac;
      uint2 p;
      p.x = pack2(src[0], src[1]);
      p.y = pack2(src[2], src[3]);
      *(uint2*)(As + (ar >> 4) * 512 + a_off(ar & 15, ac)) = p;
    }
    {   // B tile (activations): 8 coalesced dword loads -> uint4 store
      const float* src = xbase + (size_t)(k0 + kq * 8) * HW_ + ng;
      float v0 = src[0 * HW_], v1 = src[1 * HW_], v2 = src[2 * HW_], v3 = src[3 * HW_];
      float v4 = src[4 * HW_], v5 = src[5 * HW_], v6 = src[6 * HW_], v7 = src[7 * HW_];
      uint4 p;
      p.x = pack2(v0, v1); p.y = pack2(v2, v3);
      p.z = pack2(v4, v5); p.w = pack2(v6, v7);
      *(uint4*)(Bs + (ng >> 4) * 512 + b_off(kq * 8, ng & 15)) = p;
    }
    if (k0 + 32 < C_) __builtin_prefetch(xbase + (size_t)(k0 + 32) * HW_, 0, 0);
    __syncthreads();
    v16bf a  = frag_ld(As + msub * 512, lane);
    v16bf bb = frag_ld(Bs + nsub * 512, lane);
    acc = wmma_bf16(a, bb, acc);
    __syncthreads();
  }
  const int n = lane & 15, half = lane >> 4;
#pragma unroll
  for (int r = 0; r < 8; ++r) {
    const int co = co0 + msub * 16 + half * 8 + r;
    const int wc = w0 + nsub * 16 + n;
    const int nh = co / HD_, hd = co % HD_;
    const size_t base = (((size_t)b * NH_ + nh) * H_ + h) * 12288;
    const unsigned short val = f2bf(acc[r] + bias[co]);
    if (MODE == 0)      y[q_idx(base, wc, hd)] = val;
    else if (MODE == 1) y[k_idx(base, wc, hd)] = val;
    else                y[v_idx(base, wc, hd)] = val;
  }
}

// ---------------------------------------------------------------------------
// Kernel 2: per-row attention.  q/k/v arrive fragment-major -> straight uint4
// LDS copies; every WMMA operand is a 2x ds_load_b128 fragment read.
// Output (bf16) goes to a pos-major [B,H,W,C] buffer for the next GEMMs.
// ---------------------------------------------------------------------------
__global__ __launch_bounds__(256) void k_attn(const unsigned short* __restrict__ qg,
                                              const unsigned short* __restrict__ kg,
                                              const unsigned short* __restrict__ vg,
                                              unsigned short* __restrict__ og) {
  __shared__ __align__(16) unsigned short qs[12288];   // A-frags [8][3][512]
  __shared__ __align__(16) unsigned short kv[12288];   // k then v frags
  __shared__ __align__(16) unsigned short Ps[16384];   // P A-frags [8][4][512]
  const int tid = threadIdx.x, lane = tid & 31, wid = tid >> 5;
  const size_t base = (size_t)blockIdx.x * 12288;
  const int b = blockIdx.x >> 8, nh = (blockIdx.x >> 6) & 3, h = blockIdx.x & 63;

  {   // load q, k (1536 uint4 each)
    const uint4* sq = (const uint4*)(qg + base);
    const uint4* sk = (const uint4*)(kg + base);
    uint4* dq = (uint4*)qs;
    uint4* dk = (uint4*)kv;
#pragma unroll
    for (int i = 0; i < 6; ++i) { dq[tid + 256 * i] = sq[tid + 256 * i]; dk[tid + 256 * i] = sk[tid + 256 * i]; }
  }
  __syncthreads();

  // Phase 1: S strip (16 w rows) x 128 cols
  v8f acc[8];
#pragma unroll
  for (int t = 0; t < 8; ++t) { v8f z = {}; acc[t] = z; }
#pragma unroll
  for (int kc = 0; kc < 3; ++kc) {
    v16bf a = frag_ld(qs + wid * 1536 + kc * 512, lane);
#pragma unroll
    for (int t = 0; t < 8; ++t) {
      v16bf bb = frag_ld(kv + (t * 3 + kc) * 512, lane);
      acc[t] = wmma_bf16(a, bb, acc[t]);
    }
  }

  const float scale = 0.10206207261596577f;   // 1/sqrt(96)
  const int n = lane & 15, half = lane >> 4;
#pragma unroll
  for (int r = 0; r < 8; ++r) {
    float mx = -3.0e38f;
#pragma unroll
    for (int t = 0; t < 8; ++t) { float v = acc[t][r] * scale; acc[t][r] = v; mx = fmaxf(mx, v); }
    mx = fmaxf(mx, __shfl_xor(mx, 1));
    mx = fmaxf(mx, __shfl_xor(mx, 2));
    mx = fmaxf(mx, __shfl_xor(mx, 4));
    mx = fmaxf(mx, __shfl_xor(mx, 8));
    float sum = 0.f;
#pragma unroll
    for (int t = 0; t < 8; ++t) { float e = __expf(acc[t][r] - mx); acc[t][r] = e; sum += e; }
    sum += __shfl_xor(sum, 1);
    sum += __shfl_xor(sum, 2);
    sum += __shfl_xor(sum, 4);
    sum += __shfl_xor(sum, 8);
    const float inv = 1.f / sum;
    const int m = half * 8 + r;               // row within strip
#pragma unroll
    for (int t = 0; t < 8; ++t) {             // P in A-fragment order
      const int col = t * 16 + n;
      Ps[wid * 2048 + (col >> 5) * 512 + a_off(m, col & 31)] = f2bf(acc[t][r] * inv);
    }
  }
  __syncthreads();

  {   // load v over k's LDS slot (fragment-major already)
    const uint4* sv = (const uint4*)(vg + base);
    uint4* dv = (uint4*)kv;
#pragma unroll
    for (int i = 0; i < 6; ++i) dv[tid + 256 * i] = sv[tid + 256 * i];
  }
  __syncthreads();

  // Phase 2: O strip = P[strip] x V  -> [16 w][96 hd]
  v8f acc2[6];
#pragma unroll
  for (int t = 0; t < 6; ++t) { v8f z = {}; acc2[t] = z; }
#pragma unroll
  for (int kc = 0; kc < 4; ++kc) {
    v16bf a = frag_ld(Ps + wid * 2048 + kc * 512, lane);
#pragma unroll
    for (int t = 0; t < 6; ++t) {
      v16bf bb = frag_ld(kv + (t * 4 + kc) * 512, lane);
      acc2[t] = wmma_bf16(a, bb, acc2[t]);
    }
  }
  // write pos-major [B,H,W,C] bf16, c = nh*96 + hd
#pragma unroll
  for (int t = 0; t < 6; ++t)
#pragma unroll
    for (int r = 0; r < 8; ++r) {
      const int row = wid * 16 + half * 8 + r;
      const size_t oi = (((size_t)b * H_ + h) * W_ + row) * C_ + nh * HD_ + t * 16 + n;
      og[oi] = f2bf(acc2[t][r]);
    }
}

// ---------------------------------------------------------------------------
// Kernel 3: output projection.  obuf[pos][co] = bf16( Wo . attn + bo ).
// B tiles load contiguously (uint4) from the pos-major attention buffer.
// ---------------------------------------------------------------------------
__global__ __launch_bounds__(256) void k_outproj(const unsigned short* __restrict__ ag,
                                                 const float* __restrict__ wo,
                                                 const float* __restrict__ bo,
                                                 unsigned short* __restrict__ obuf) {
  __shared__ __align__(16) unsigned short As[2 * 512];
  __shared__ __align__(16) unsigned short Bs[4 * 512];
  const int tid = threadIdx.x, lane = tid & 31, wid = tid >> 5;
  const int msub = wid >> 2, nsub = wid & 3;
  const int pos0 = blockIdx.x * 64;
  const int co0  = blockIdx.y * 32;

  const int ar = tid >> 3, ac = (tid & 7) * 4;
  const int kq = tid >> 6, ng = tid & 63;

  v8f acc = {};
  for (int k0 = 0; k0 < C_; k0 += 32) {
    {
      const float* src = wo + (size_t)(co0 + ar) * C_ + k0 + ac;
      uint2 p;
      p.x = pack2(src[0], src[1]);
      p.y = pack2(src[2], src[3]);
      *(uint2*)(As + (ar >> 4) * 512 + a_off(ar & 15, ac)) = p;
    }
    {   // contiguous 8 channels for one position
      const uint4 p = *(const uint4*)(ag + (size_t)(pos0 + ng) * C_ + k0 + kq * 8);
      *(uint4*)(Bs + (ng >> 4) * 512 + b_off(kq * 8, ng & 15)) = p;
    }
    __syncthreads();
    v16bf a  = frag_ld(As + msub * 512, lane);
    v16bf bb = frag_ld(Bs + nsub * 512, lane);
    acc = wmma_bf16(a, bb, acc);
    __syncthreads();
  }
  const int n = lane & 15, half = lane >> 4;
#pragma unroll
  for (int r = 0; r < 8; ++r) {
    const int co = co0 + msub * 16 + half * 8 + r;
    const int pos = pos0 + nsub * 16 + n;
    obuf[(size_t)pos * C_ + co] = f2bf(acc[r] + bo[co]);
  }
}

// ---------------------------------------------------------------------------
// Kernel 4: gate GEMM (K=768: fp32 feat_self then bf16 Wo output), fused
// sigmoid gate + residual -> d_out (fp32, [B,C,H,W]).
// ---------------------------------------------------------------------------
__global__ __launch_bounds__(256) void k_gatefuse(const float* __restrict__ xself,
                                                  const unsigned short* __restrict__ obuf,
                                                  const float* __restrict__ wg,
                                                  const float* __restrict__ bg,
                                                  float* __restrict__ out) {
  __shared__ __align__(16) unsigned short As[2 * 512];
  __shared__ __align__(16) unsigned short Bs[4 * 512];
  const int tid = threadIdx.x, lane = tid & 31, wid = tid >> 5;
  const int msub = wid >> 2, nsub = wid & 3;
  const int pos0 = blockIdx.x * 64;
  const int co0  = blockIdx.y * 32;
  const int b = pos0 >> 13, rem = pos0 & 8191, h = rem >> 7, w0 = rem & 127;
  const float* xbase = xself + (size_t)b * C_ * HW_ + h * W_ + w0;

  const int ar = tid >> 3, ac = (tid & 7) * 4;
  const int kq = tid >> 6, ng = tid & 63;

  v8f acc = {};
  for (int k0 = 0; k0 < 2 * C_; k0 += 32) {
    {
      const float* src = wg + (size_t)(co0 + ar) * (2 * C_) + k0 + ac;
      uint2 p;
      p.x = pack2(src[0], src[1]);
      p.y = pack2(src[2], src[3]);
      *(uint2*)(As + (ar >> 4) * 512 + a_off(ar & 15, ac)) = p;
    }
    if (k0 < C_) {   // fp32 feat_self half
      const float* src = xbase + (size_t)(k0 + kq * 8) * HW_ + ng;
      float v0 = src[0 * HW_], v1 = src[1 * HW_], v2 = src[2 * HW_], v3 = src[3 * HW_];
      float v4 = src[4 * HW_], v5 = src[5 * HW_], v6 = src[6 * HW_], v7 = src[7 * HW_];
      uint4 p;
      p.x = pack2(v0, v1); p.y = pack2(v2, v3);
      p.z = pack2(v4, v5); p.w = pack2(v6, v7);
      *(uint4*)(Bs + (ng >> 4) * 512 + b_off(kq * 8, ng & 15)) = p;
    } else {          // bf16 Wo-output half (pos-major, contiguous channels)
      const uint4 p = *(const uint4*)(obuf + (size_t)(pos0 + ng) * C_ + (k0 - C_) + kq * 8);
      *(uint4*)(Bs + (ng >> 4) * 512 + b_off(kq * 8, ng & 15)) = p;
    }
    __syncthreads();
    v16bf a  = frag_ld(As + msub * 512, lane);
    v16bf bb = frag_ld(Bs + nsub * 512, lane);
    acc = wmma_bf16(a, bb, acc);
    __syncthreads();
  }
  const int n = lane & 15, half = lane >> 4;
#pragma unroll
  for (int r = 0; r < 8; ++r) {
    const int co = co0 + msub * 16 + half * 8 + r;
    const int wc = w0 + nsub * 16 + n;
    const int pos = pos0 + nsub * 16 + n;
    const size_t fi = ((size_t)b * C_ + co) * HW_ + h * W_ + wc;
    float g = acc[r] + bg[co];
    g = 1.f / (1.f + __expf(-g));
    out[fi] = xself[fi] + g * bf2f(obuf[(size_t)pos * C_ + co]);
  }
}

// ---------------------------------------------------------------------------
// BatchNorm: deterministic two-stage reduction, then in-place apply.
// ---------------------------------------------------------------------------
__global__ __launch_bounds__(256) void bn_partial(const float* __restrict__ f,
                                                  float* __restrict__ ps,
                                                  float* __restrict__ psq) {
  const int c = blockIdx.x, seg = blockIdx.y, tid = threadIdx.x;
  const int f0 = seg * 1024 + tid * 4;
  float s = 0.f, sq = 0.f;
#pragma unroll
  for (int j = 0; j < 4; ++j) {
    const int fl = f0 + j;
    const int b = fl >> 13, rem = fl & 8191;
    const float v = f[((size_t)b * C_ + c) * HW_ + rem];
    s += v; sq += v * v;
  }
  __shared__ float ls[256], lq[256];
  ls[tid] = s; lq[tid] = sq;
  __syncthreads();
  for (int st = 128; st > 0; st >>= 1) {
    if (tid < st) { ls[tid] += ls[tid + st]; lq[tid] += lq[tid + st]; }
    __syncthreads();
  }
  if (tid == 0) { ps[c * 64 + seg] = ls[0]; psq[c * 64 + seg] = lq[0]; }
}

__global__ __launch_bounds__(64) void bn_finalize(const float* __restrict__ ps,
                                                  const float* __restrict__ psq,
                                                  const float* __restrict__ gamma,
                                                  const float* __restrict__ beta,
                                                  float* __restrict__ scl,
                                                  float* __restrict__ shf) {
  const int c = blockIdx.x, tid = threadIdx.x;
  __shared__ float ls[64], lq[64];
  ls[tid] = ps[c * 64 + tid];
  lq[tid] = psq[c * 64 + tid];
  __syncthreads();
  for (int st = 32; st > 0; st >>= 1) {
    if (tid < st) { ls[tid] += ls[tid + st]; lq[tid] += lq[tid + st]; }
    __syncthreads();
  }
  if (tid == 0) {
    const float mean = ls[0] * (1.f / 65536.f);
    const float var  = lq[0] * (1.f / 65536.f) - mean * mean;
    const float inv  = rsqrtf(var + 1e-5f);
    scl[c] = inv * gamma[c];
    shf[c] = beta[c] - mean * inv * gamma[c];
  }
}

__global__ __launch_bounds__(256) void bn_apply(float* __restrict__ out,
                                                const float* __restrict__ scl,
                                                const float* __restrict__ shf) {
  const size_t i = ((size_t)blockIdx.x * 256 + threadIdx.x) * 4;
  const int c = (int)((i >> 13) % C_);
  const float s = scl[c], t = shf[c];
  float4 v = *(const float4*)(out + i);
  v.x = v.x * s + t; v.y = v.y * s + t; v.z = v.z * s + t; v.w = v.w * s + t;
  *(float4*)(out + i) = v;
}

// ---------------------------------------------------------------------------
// Launch.  Workspace (~202 MB): q | k | v | attn-out (bf16) | BN partials.
// The Wo output overlays the (dead) q buffer.
// ---------------------------------------------------------------------------
extern "C" void kernel_launch(void* const* d_in, const int* in_sizes, int n_in,
                              void* d_out, int out_size, void* d_ws, size_t ws_size,
                              hipStream_t stream) {
  (void)in_sizes; (void)n_in; (void)out_size; (void)ws_size;
  const float* feat_self  = (const float*)d_in[0];
  const float* feat_other = (const float*)d_in[1];
  const float* wq = (const float*)d_in[2];
  const float* bq = (const float*)d_in[3];
  const float* wk = (const float*)d_in[4];
  const float* bk = (const float*)d_in[5];
  const float* wv = (const float*)d_in[6];
  const float* bv = (const float*)d_in[7];
  const float* wo = (const float*)d_in[8];
  const float* bo = (const float*)d_in[9];
  const float* wg = (const float*)d_in[10];
  const float* bg = (const float*)d_in[11];
  const float* gamma = (const float*)d_in[12];
  const float* beta  = (const float*)d_in[13];
  float* out = (float*)d_out;

  unsigned short* qbuf = (unsigned short*)d_ws;
  unsigned short* kbuf = qbuf + NQ;
  unsigned short* vbuf = kbuf + NQ;
  unsigned short* og   = vbuf + NQ;        // attention out, pos-major [B,H,W,C]
  unsigned short* obuf = qbuf;             // Wo output overlays dead q buffer
  float* ps  = (float*)(og + NX);
  float* psq = ps + C_ * 64;
  float* scl = psq + C_ * 64;
  float* shf = scl + C_;

  const dim3 gProj(NPOS / 64, C_ / 32);
  k_proj<0><<<gProj, 256, 0, stream>>>(feat_self,  wq, bq, qbuf);
  k_proj<1><<<gProj, 256, 0, stream>>>(feat_other, wk, bk, kbuf);
  k_proj<2><<<gProj, 256, 0, stream>>>(feat_other, wv, bv, vbuf);
  k_attn<<<B_ * NH_ * H_, 256, 0, stream>>>(qbuf, kbuf, vbuf, og);
  k_outproj<<<gProj, 256, 0, stream>>>(og, wo, bo, obuf);
  k_gatefuse<<<gProj, 256, 0, stream>>>(feat_self, obuf, wg, bg, out);
  bn_partial<<<dim3(C_, 64), 256, 0, stream>>>(out, ps, psq);
  bn_finalize<<<C_, 64, 0, stream>>>(ps, psq, gamma, beta, scl, shf);
  bn_apply<<<(unsigned)(NX / 1024), 256, 0, stream>>>(out, scl, shf);
}
